// TransformerEncoder_61942018343246
// MI455X (gfx1250) — compile-verified, matrix operand order checked by reading, not run on hardware
//
#include <hip/hip_runtime.h>

// ---------------- constants ----------------
constexpr int Bn  = 4;
constexpr int Sn  = 2048;
constexpr int Dm  = 256;
constexpr int Hn  = 8;
constexpr int DKn = 32;     // Dm / Hn
constexpr int FFn = 1024;   // 4 * Dm
constexpr float LN_EPS = 1e-5f;
constexpr float QK_SCALE = 0.17677669529663687f; // 1/sqrt(32)

// ---------------- WMMA types ----------------
typedef __attribute__((ext_vector_type(16))) __bf16 v16bf;
typedef __attribute__((ext_vector_type(8)))  float  v8f;

union BF16x16 { v16bf v; unsigned short u[16]; uint4 q2[2]; };
union US8 { unsigned short u[8]; uint4 q; };

// native f32 -> bf16 (RNE) via hardware convert
__device__ __forceinline__ unsigned short f2bf(float x) {
  union { __bf16 b; unsigned short u; } c;
  c.b = (__bf16)x;
  return c.u;
}

__device__ __forceinline__ v8f wmma_bf16(v16bf a, v16bf b, v8f c) {
  return __builtin_amdgcn_wmma_f32_16x16x32_bf16(false, a, false, b, (short)0, c,
                                                 false, false);
}

// ---- operand loaders (16x16x32 bf16 WMMA, wave32 layouts) ----
// A tile (rows m0.., k chunk k0..) from bf16 row-major, ld in elements.
__device__ __forceinline__ v16bf load_a_bf16(const unsigned short* A, int ld, int m0, int k0, int lane) {
  int r  = m0 + (lane & 15);
  int kk = k0 + ((lane < 16) ? 0 : 8);
  const unsigned short* p = A + (size_t)r * ld + kk;
  BF16x16 o;
  o.q2[0] = *(const uint4*)(p);
  o.q2[1] = *(const uint4*)(p + 16);
  return o.v;
}
// B tile from bf16 stored "B-layout": element (k,n) at n*ld + k  (contiguous in k)
__device__ __forceinline__ v16bf load_bT(const unsigned short* Bm, int ld, int k0, int n0, int lane) {
  int n  = n0 + (lane & 15);
  int kk = k0 + ((lane < 16) ? 0 : 8);
  const unsigned short* p = Bm + (size_t)n * ld + kk;
  BF16x16 o;
  o.q2[0] = *(const uint4*)(p);
  o.q2[1] = *(const uint4*)(p + 16);
  return o.v;
}
// A from per-wave LDS bf16 patch (16 rows x 32 cols)
__device__ __forceinline__ v16bf load_a_lds(const unsigned short* p0, int lane) {
  const unsigned short* p = p0 + (lane & 15) * 32 + ((lane < 16) ? 0 : 8);
  BF16x16 o;
  o.q2[0] = *(const uint4*)(p);
  o.q2[1] = *(const uint4*)(p + 16);
  return o.v;
}

// ================= cast kernels (one-time f32 -> bf16) =================
__global__ __launch_bounds__(256) void cast_x_kernel(const float* __restrict__ x,
                                                     unsigned short* __restrict__ xh, int n) {
  int i = (blockIdx.x * 256 + threadIdx.x) * 8;
  if (i + 8 <= n) {
    float4 a = *(const float4*)(x + i);
    float4 b = *(const float4*)(x + i + 4);
    US8 o;
    o.u[0] = f2bf(a.x); o.u[1] = f2bf(a.y); o.u[2] = f2bf(a.z); o.u[3] = f2bf(a.w);
    o.u[4] = f2bf(b.x); o.u[5] = f2bf(b.y); o.u[6] = f2bf(b.z); o.u[7] = f2bf(b.w);
    *(uint4*)(xh + i) = o.q;
  }
}
// W[K][N] f32 row-major  ->  Wt bf16 with element (k,n) at n*K + k
__global__ __launch_bounds__(256) void castT_w_kernel(const float* __restrict__ W,
                                                      unsigned short* __restrict__ Wt,
                                                      int K, int N) {
  int idx = blockIdx.x * 256 + threadIdx.x;
  if (idx < K * N) {
    int k = idx / N, n = idx % N;
    Wt[(size_t)n * K + k] = f2bf(W[idx]);
  }
}

// ================= kernel 1: fused QKV projection =================
// q: bf16 [B][H][S][DK] (pre-scaled by 1/sqrt(DK)); k: [B][H][S][DK]; v: TRANSPOSED [B][H][DK][S]
__global__ __launch_bounds__(256) void qkv_kernel(const unsigned short* __restrict__ xh,
                                                  const unsigned short* __restrict__ WqT,
                                                  const unsigned short* __restrict__ WkT,
                                                  const unsigned short* __restrict__ WvT,
                                                  unsigned short* __restrict__ qo,
                                                  unsigned short* __restrict__ ko,
                                                  unsigned short* __restrict__ vT) {
  int lane = threadIdx.x & 31;
  int wave = __builtin_amdgcn_readfirstlane(threadIdx.x >> 5);
  int m0 = blockIdx.x * 16;
  int b  = m0 / Sn;            // whole slab is in one batch (Sn % 16 == 0)
  int s0 = m0 % Sn;
  int mb = (lane < 16) ? 0 : 8;
  int nc = lane & 15;

#pragma unroll
  for (int which = 0; which < 3; ++which) {
    const unsigned short* Wt = (which == 0) ? WqT : ((which == 1) ? WkT : WvT);
    for (int t = wave; t < 16; t += 8) {       // 2 n-tiles per wave
      int n0 = t * 16;
      v8f acc = {};
#pragma unroll
      for (int k0 = 0; k0 < Dm; k0 += 32) {
        acc = wmma_bf16(load_a_bf16(xh, Dm, m0, k0, lane),
                        load_bT(Wt, Dm, k0, n0, lane), acc);
      }
      int n  = n0 + nc;
      int h  = n >> 5, d = n & 31;
      int bh = b * Hn + h;
      if (which == 0) {
        unsigned short* p = qo + ((size_t)bh * Sn + s0 + mb) * DKn + d;
#pragma unroll
        for (int r = 0; r < 8; ++r) p[(size_t)r * DKn] = f2bf(acc[r] * QK_SCALE);
      } else if (which == 1) {
        unsigned short* p = ko + ((size_t)bh * Sn + s0 + mb) * DKn + d;
#pragma unroll
        for (int r = 0; r < 8; ++r) p[(size_t)r * DKn] = f2bf(acc[r]);
      } else {
        unsigned short* p = vT + ((size_t)bh * DKn + d) * Sn + s0 + mb;
#pragma unroll
        for (int r = 0; r < 8; ++r) p[r] = f2bf(acc[r]);
      }
    }
  }
}

// ================= kernel 2: flash attention (one q-tile per wave) =================
// attn out: bf16 [B*S][D] (heads re-concatenated)
__global__ __launch_bounds__(256) void attn_kernel(const unsigned short* __restrict__ q,
                                                   const unsigned short* __restrict__ k,
                                                   const unsigned short* __restrict__ vT,
                                                   unsigned short* __restrict__ attn) {
  __shared__ unsigned short pstage[8][16 * 32];   // per-wave bf16 P staging, 8 KB
  int lane = threadIdx.x & 31;
  int wave = __builtin_amdgcn_readfirstlane(threadIdx.x >> 5);
  int groups_per_bh = (Sn / 16) / 8;     // 16
  int bh  = blockIdx.x / groups_per_bh;
  int grp = blockIdx.x % groups_per_bh;
  int qt  = grp * 8 + wave;
  int b = bh / Hn, h = bh % Hn;
  const unsigned short* qb = q  + (size_t)bh * Sn * DKn;
  const unsigned short* kb = k  + (size_t)bh * Sn * DKn;
  const unsigned short* vb = vT + (size_t)bh * DKn * Sn;
  int q0 = qt * 16;

  v16bf aq = load_a_bf16(qb, DKn, q0, 0, lane);
  v8f o0 = {}, o1 = {};
  float rm[8], rs[8];
#pragma unroll
  for (int r = 0; r < 8; ++r) { rm[r] = -3.0e38f; rs[r] = 0.0f; }
  unsigned short* myp = &pstage[wave][0];
  int mb = (lane < 16) ? 0 : 8;
  int nc = lane & 15;

  for (int st = 0; st < Sn / 16; st += 2) {   // 32 keys per step
    // scores: K stored [s][dk] == B-layout (n=s, k=dk contiguous)
    v16bf bk0 = load_bT(kb, DKn, 0, st * 16, lane);
    v16bf bk1 = load_bT(kb, DKn, 0, st * 16 + 16, lane);
    v8f z = {};
    v8f c0 = wmma_bf16(aq, bk0, z);
    v8f c1 = wmma_bf16(aq, bk1, z);

    float p0[8], p1[8];
#pragma unroll
    for (int r = 0; r < 8; ++r) {
      float t = fmaxf(c0[r], c1[r]);
      t = fmaxf(t, __shfl_xor(t, 1, 16));
      t = fmaxf(t, __shfl_xor(t, 2, 16));
      t = fmaxf(t, __shfl_xor(t, 4, 16));
      t = fmaxf(t, __shfl_xor(t, 8, 16));
      float nm  = fmaxf(rm[r], t);
      float fac = __expf(rm[r] - nm);
      rm[r] = nm;
      p0[r] = __expf(c0[r] - nm);
      p1[r] = __expf(c1[r] - nm);
      rs[r] = rs[r] * fac + p0[r] + p1[r];
      o0[r] *= fac;
      o1[r] *= fac;
    }
    // stage unnormalized probs (bf16 16x32) to per-wave LDS, re-read in A layout
#pragma unroll
    for (int r = 0; r < 8; ++r) {
      myp[(mb + r) * 32 + nc]      = f2bf(p0[r]);
      myp[(mb + r) * 32 + 16 + nc] = f2bf(p1[r]);
    }
    asm volatile("s_wait_dscnt 0" ::: "memory");   // in-wave LDS RAW
    v16bf ap  = load_a_lds(myp, lane);
    // PV: vT stored [dk][s] == B-layout (n=dk, k=s contiguous)
    v16bf bv0 = load_bT(vb, Sn, st * 16, 0, lane);
    v16bf bv1 = load_bT(vb, Sn, st * 16, 16, lane);
    o0 = wmma_bf16(ap, bv0, o0);
    o1 = wmma_bf16(ap, bv1, o1);
  }
  // reduce row sums across 16 lanes, normalize, store
#pragma unroll
  for (int r = 0; r < 8; ++r) {
    float s = rs[r];
    s += __shfl_xor(s, 1, 16);
    s += __shfl_xor(s, 2, 16);
    s += __shfl_xor(s, 4, 16);
    s += __shfl_xor(s, 8, 16);
    float inv = 1.0f / s;
    int row = b * Sn + q0 + mb + r;
    attn[(size_t)row * Dm + h * DKn + nc]      = f2bf(o0[r] * inv);
    attn[(size_t)row * Dm + h * DKn + 16 + nc] = f2bf(o1[r] * inv);
  }
}

// ================= kernel 3/5: GEMM (bf16 A, bf16 B-layout W) + residual + LayerNorm =================
// out[m] = LN(resid[m] + Ah[m] @ W);   Wt: bf16, element (k,n) at n*K + k
template <int K>
__global__ __launch_bounds__(256) void gemm_res_ln_kernel(const unsigned short* __restrict__ Ah,
                                                          const unsigned short* __restrict__ Wt,
                                                          const float* __restrict__ resid,
                                                          const float* __restrict__ g,
                                                          const float* __restrict__ bt,
                                                          float* __restrict__ outF,
                                                          unsigned short* __restrict__ outH) {
  __shared__ float tile[16 * Dm];
  __shared__ float partS[256], partQ[256];
  __shared__ float mu_s[16], rstd_s[16];
  int lane = threadIdx.x & 31;
  int wave = __builtin_amdgcn_readfirstlane(threadIdx.x >> 5);
  int m0 = blockIdx.x * 16;

  for (int t = wave; t < Dm / 16; t += 8) {
    int n0 = t * 16;
    v8f acc = {};
#pragma unroll
    for (int k0 = 0; k0 < K; k0 += 32) {
      acc = wmma_bf16(load_a_bf16(Ah, K, m0, k0, lane),
                      load_bT(Wt, K, k0, n0, lane), acc);
    }
    int mb = (lane < 16) ? 0 : 8;
    int nc = lane & 15;
#pragma unroll
    for (int r = 0; r < 8; ++r) tile[(mb + r) * Dm + n0 + nc] = acc[r];
  }
  __syncthreads();

  int r    = threadIdx.x >> 4;       // row 0..15
  int cseg = threadIdx.x & 15;       // 16-col segment
  float sm = 0.f, sq = 0.f;
#pragma unroll
  for (int j = 0; j < 16; ++j) {
    int col = cseg * 16 + j;
    float val = tile[r * Dm + col] + resid[(size_t)(m0 + r) * Dm + col];
    tile[r * Dm + col] = val;
    sm += val; sq += val * val;
  }
  partS[r * 16 + cseg] = sm;
  partQ[r * 16 + cseg] = sq;
  __syncthreads();
  if (threadIdx.x < 16) {
    float s = 0.f, q2 = 0.f;
#pragma unroll
    for (int j = 0; j < 16; ++j) { s += partS[threadIdx.x * 16 + j]; q2 += partQ[threadIdx.x * 16 + j]; }
    float mu  = s / Dm;
    float var = q2 / Dm - mu * mu;
    mu_s[threadIdx.x]   = mu;
    rstd_s[threadIdx.x] = rsqrtf(var + LN_EPS);
  }
  __syncthreads();
  float mu = mu_s[r], rstd = rstd_s[r];
#pragma unroll
  for (int j = 0; j < 16; ++j) {
    int col = cseg * 16 + j;
    float y = (tile[r * Dm + col] - mu) * rstd * g[col] + bt[col];
    size_t a = (size_t)(m0 + r) * Dm + col;
    outF[a] = y;
    if (outH) outH[a] = f2bf(y);
  }
}

// ================= kernel 4: GLU feed-forward  h = silu(x1@W1) * (x1@W2) =================
__global__ __launch_bounds__(256) void ffn_glu_kernel(const unsigned short* __restrict__ x1h,
                                                      const unsigned short* __restrict__ W1T,
                                                      const unsigned short* __restrict__ W2T,
                                                      unsigned short* __restrict__ hbuf) {
  int lane = threadIdx.x & 31;
  int wave = __builtin_amdgcn_readfirstlane(threadIdx.x >> 5);
  int m0 = blockIdx.x * 16;
  int mb = (lane < 16) ? 0 : 8;
  int nc = lane & 15;
  for (int t = wave; t < FFn / 16; t += 8) {
    int n0 = t * 16;
    v8f a1 = {}, a2 = {};
#pragma unroll
    for (int k0 = 0; k0 < Dm; k0 += 32) {
      v16bf a = load_a_bf16(x1h, Dm, m0, k0, lane);
      a1 = wmma_bf16(a, load_bT(W1T, Dm, k0, n0, lane), a1);
      a2 = wmma_bf16(a, load_bT(W2T, Dm, k0, n0, lane), a2);
    }
    unsigned short* p = hbuf + (size_t)(m0 + mb) * FFn + n0 + nc;
#pragma unroll
    for (int r = 0; r < 8; ++r) {
      float w = a1[r];
      float sil = w / (1.0f + __expf(-w));
      p[(size_t)r * FFn] = f2bf(sil * a2[r]);
    }
  }
}

// ================= host launch =================
extern "C" void kernel_launch(void* const* d_in, const int* in_sizes, int n_in,
                              void* d_out, int out_size, void* d_ws, size_t ws_size,
                              hipStream_t stream) {
  (void)in_sizes; (void)n_in; (void)out_size; (void)ws_size;
  const float* x    = (const float*)d_in[0];
  const float* Wq   = (const float*)d_in[1];
  const float* Wk   = (const float*)d_in[2];
  const float* Wv   = (const float*)d_in[3];
  const float* Wo   = (const float*)d_in[4];
  const float* W1   = (const float*)d_in[5];
  const float* W2   = (const float*)d_in[6];
  const float* Wout = (const float*)d_in[7];
  const float* g1   = (const float*)d_in[8];
  const float* b1   = (const float*)d_in[9];
  const float* g2   = (const float*)d_in[10];
  const float* b2   = (const float*)d_in[11];

  const size_t M = (size_t)Bn * Sn;                 // 8192 rows
  char* ws = (char*)d_ws;
  auto take = [&](size_t bytes) { char* p = ws; ws += (bytes + 255) & ~(size_t)255; return p; };

  unsigned short* xh   = (unsigned short*)take(M * Dm * 2);
  unsigned short* qbuf = (unsigned short*)take(M * Dm * 2);
  unsigned short* kbuf = (unsigned short*)take(M * Dm * 2);
  unsigned short* vTb  = (unsigned short*)take(M * Dm * 2);
  unsigned short* attn = (unsigned short*)take(M * Dm * 2);
  unsigned short* x1h  = (unsigned short*)take(M * Dm * 2);
  unsigned short* hbuf = (unsigned short*)take(M * FFn * 2);
  float*          x1f  = (float*)take(M * Dm * 4);
  unsigned short* WqT  = (unsigned short*)take((size_t)Dm * Dm * 2);
  unsigned short* WkT  = (unsigned short*)take((size_t)Dm * Dm * 2);
  unsigned short* WvT  = (unsigned short*)take((size_t)Dm * Dm * 2);
  unsigned short* WoT  = (unsigned short*)take((size_t)Dm * Dm * 2);
  unsigned short* W1T  = (unsigned short*)take((size_t)Dm * FFn * 2);
  unsigned short* W2T  = (unsigned short*)take((size_t)Dm * FFn * 2);
  unsigned short* WoutT= (unsigned short*)take((size_t)FFn * Dm * 2);

  // one-time casts (tiny vs. the GEMM work)
  cast_x_kernel<<<(int)(M * Dm / 8 / 256), 256, 0, stream>>>(x, xh, (int)(M * Dm));
  castT_w_kernel<<<(Dm * Dm + 255) / 256, 256, 0, stream>>>(Wq, WqT, Dm, Dm);
  castT_w_kernel<<<(Dm * Dm + 255) / 256, 256, 0, stream>>>(Wk, WkT, Dm, Dm);
  castT_w_kernel<<<(Dm * Dm + 255) / 256, 256, 0, stream>>>(Wv, WvT, Dm, Dm);
  castT_w_kernel<<<(Dm * Dm + 255) / 256, 256, 0, stream>>>(Wo, WoT, Dm, Dm);
  castT_w_kernel<<<(Dm * FFn + 255) / 256, 256, 0, stream>>>(W1, W1T, Dm, FFn);
  castT_w_kernel<<<(Dm * FFn + 255) / 256, 256, 0, stream>>>(W2, W2T, Dm, FFn);
  castT_w_kernel<<<(FFn * Dm + 255) / 256, 256, 0, stream>>>(Wout, WoutT, FFn, Dm);

  const int slabs = (int)(M / 16);                  // 512
  qkv_kernel<<<slabs, 256, 0, stream>>>(xh, WqT, WkT, WvT, qbuf, kbuf, vTb);
  attn_kernel<<<Bn * Hn * ((Sn / 16) / 8), 256, 0, stream>>>(qbuf, kbuf, vTb, attn);
  gemm_res_ln_kernel<Dm><<<slabs, 256, 0, stream>>>(attn, WoT, x, g1, b1, x1f, x1h);
  ffn_glu_kernel<<<slabs, 256, 0, stream>>>(x1h, W1T, W2T, hbuf);
  gemm_res_ln_kernel<FFn><<<slabs, 256, 0, stream>>>(hbuf, WoutT, x1f, g2, b2,
                                                     (float*)d_out, nullptr);
}
